// RNN_88124138979567
// MI455X (gfx1250) — matrix operand order repaired and checked by reading, not compiled, not run on hardware
//
#include <hip/hip_runtime.h>
#include <hip/hip_bf16.h>

// ---------------------------------------------------------------------------
// 2-layer tanh RNN on MI455X (gfx1250): per-step bf16 WMMA GEMMs, fp32 accum.
// ---------------------------------------------------------------------------

typedef __bf16 bf16_t;
typedef __attribute__((ext_vector_type(16))) __bf16 v16bf;
typedef __attribute__((ext_vector_type(8)))  __bf16 v8bf;
typedef __attribute__((ext_vector_type(8)))  float  v8f;

#define B_   512
#define T_   90
#define I_   1024
#define H1_  2048
#define H2_  2048
#define K1CAT (I_ + H1_)   // 3072
#define K2CAT (H1_ + H2_)  // 4096

// --------------------------- small helper kernels --------------------------

// dst[r*ldd + c] = (bf16) src[r*lds + c]
__global__ void cvt_rows_kernel(const float* __restrict__ src, long lds,
                                bf16_t* __restrict__ dst, long ldd, int ncols) {
  int c = blockIdx.x * blockDim.x + threadIdx.x;
  long r = blockIdx.y;
  if (c < ncols) dst[r * ldd + c] = (bf16_t)src[r * lds + c];
}

__global__ void add_bias_kernel(const float* __restrict__ a,
                                const float* __restrict__ b,
                                float* __restrict__ o, int n) {
  int i = blockIdx.x * blockDim.x + threadIdx.x;
  if (i < n) o[i] = a[i] + b[i];
}

// ------------------------------- GEMM core ---------------------------------
// D = act( [A0 | A1] @ W^T + bias ),  W row-major [N, K0+K1] bf16.
// One wave computes a 32x32 tile (2x2 WMMA 16x16 tiles). Block = 8 waves
// covering 64(M) x 128(N). M is fixed at 512 and N is a multiple of 128.

__device__ __forceinline__ void mma_seg(const bf16_t* __restrict__ A, long lda,
                                        int Kseg,
                                        const bf16_t* __restrict__ Wp, long ldw,
                                        int m_base, int n_base, int lr, int g,
                                        v8f acc[2][2]) {
  if (Kseg <= 0) return;
  // A fragment (16-bit 16x32 layout): lane holds row m_base+lr (+16 for tile1),
  // k = 16g .. 16g+15 (contiguous) of each 32-wide K block.
  const bf16_t* ar0 = A + (long)(m_base + lr) * lda + 16 * g;
  const bf16_t* ar1 = ar0 + 16 * lda;
  // B fragment: column n = n_base+lr (+16), k = kb + 16g .. kb + 16g + 15,
  // i.e. 16 contiguous bf16 of row n of W (W is [N,K] row-major).
  const bf16_t* wr0 = Wp + (long)(n_base + lr) * ldw + 16 * g;
  const bf16_t* wr1 = wr0 + 16 * ldw;
  for (int kb = 0; kb < Kseg; kb += 32) {
    v8bf a0lo = *(const v8bf*)(ar0 + kb);
    v8bf a0hi = *(const v8bf*)(ar0 + kb + 8);
    v8bf a1lo = *(const v8bf*)(ar1 + kb);
    v8bf a1hi = *(const v8bf*)(ar1 + kb + 8);
    v16bf a0 = __builtin_shufflevector(a0lo, a0hi, 0, 1, 2, 3, 4, 5, 6, 7, 8, 9,
                                       10, 11, 12, 13, 14, 15);
    v16bf a1 = __builtin_shufflevector(a1lo, a1hi, 0, 1, 2, 3, 4, 5, 6, 7, 8, 9,
                                       10, 11, 12, 13, 14, 15);
    v16bf b0 = *(const v16bf*)(wr0 + kb);
    v16bf b1 = *(const v16bf*)(wr1 + kb);
    acc[0][0] = __builtin_amdgcn_wmma_f32_16x16x32_bf16(
        false, a0, false, b0, (short)0, acc[0][0], false, false);
    acc[0][1] = __builtin_amdgcn_wmma_f32_16x16x32_bf16(
        false, a0, false, b1, (short)0, acc[0][1], false, false);
    acc[1][0] = __builtin_amdgcn_wmma_f32_16x16x32_bf16(
        false, a1, false, b0, (short)0, acc[1][0], false, false);
    acc[1][1] = __builtin_amdgcn_wmma_f32_16x16x32_bf16(
        false, a1, false, b1, (short)0, acc[1][1], false, false);
  }
}

__global__ __launch_bounds__(256) void gemm_bf16_kernel(
    const bf16_t* __restrict__ A0, long lda0, int K0,
    const bf16_t* __restrict__ A1, long lda1, int K1,
    const bf16_t* __restrict__ W, long ldw, const float* __restrict__ bias,
    bf16_t* __restrict__ Dbf, long lddb, float* __restrict__ Df, long lddf,
    int do_tanh) {
  const int wave = threadIdx.x >> 5;
  const int lane = threadIdx.x & 31;
  const int g = lane >> 4;    // half-wave selector
  const int lr = lane & 15;   // row/col within 16
  const int m_base = blockIdx.y * 64 + (wave & 1) * 32;
  const int n_base = blockIdx.x * 128 + (wave >> 1) * 32;

  v8f acc[2][2];
#pragma unroll
  for (int i = 0; i < 2; ++i)
#pragma unroll
    for (int j = 0; j < 2; ++j) acc[i][j] = (v8f)(0.0f);

  mma_seg(A0, lda0, K0, W, ldw, m_base, n_base, lr, g, acc);
  mma_seg(A1, lda1, K1, W + K0, ldw, m_base, n_base, lr, g, acc);

  // Epilogue: C/D layout -> lane lr = column, VGPR r = row (r + 8g) per tile.
#pragma unroll
  for (int j = 0; j < 2; ++j) {
    const int n = n_base + j * 16 + lr;
    const float bv = bias[n];
#pragma unroll
    for (int i = 0; i < 2; ++i) {
#pragma unroll
      for (int r = 0; r < 8; ++r) {
        const int m = m_base + i * 16 + g * 8 + r;
        float v = acc[i][j][r] + bv;
        if (do_tanh) v = tanhf(v);
        Dbf[(long)m * lddb + n] = (bf16_t)v;
        if (Df) Df[(long)m * lddf + n] = v;
      }
    }
  }
}

// ------------------------------- host driver -------------------------------

extern "C" void kernel_launch(void* const* d_in, const int* in_sizes, int n_in,
                              void* d_out, int out_size, void* d_ws,
                              size_t ws_size, hipStream_t stream) {
  const float* x    = (const float*)d_in[0];
  const float* h1_0 = (const float*)d_in[1];
  const float* h2_0 = (const float*)d_in[2];
  const float* Wih1 = (const float*)d_in[3];
  const float* Whh1 = (const float*)d_in[4];
  const float* bih1 = (const float*)d_in[5];
  const float* bhh1 = (const float*)d_in[6];
  const float* Wih2 = (const float*)d_in[7];
  const float* Whh2 = (const float*)d_in[8];
  const float* bih2 = (const float*)d_in[9];
  const float* bhh2 = (const float*)d_in[10];
  const float* Wo   = (const float*)d_in[11];
  const float* bo   = (const float*)d_in[12];
  float* out = (float*)d_out;

  // d_out layout: outs [B,10,I] ++ h1 [B,H1] ++ h2 [B,H2]
  const size_t h1_off = (size_t)B_ * 10 * I_;
  const size_t h2_off = h1_off + (size_t)B_ * H1_;

  // workspace carve-up
  char* wp = (char*)d_ws;
  auto alloc = [&](size_t bytes) {
    char* p = wp;
    wp += (bytes + 255) & ~(size_t)255;
    return p;
  };
  bf16_t* W1cat = (bf16_t*)alloc((size_t)H1_ * K1CAT * 2);
  bf16_t* W2cat = (bf16_t*)alloc((size_t)H2_ * K2CAT * 2);
  bf16_t* Wocat = (bf16_t*)alloc((size_t)I_ * H2_ * 2);
  bf16_t* xtb   = (bf16_t*)alloc((size_t)B_ * I_ * 2);
  bf16_t* outbf = (bf16_t*)alloc((size_t)B_ * I_ * 2);
  bf16_t* h1b[2] = {(bf16_t*)alloc((size_t)B_ * H1_ * 2),
                    (bf16_t*)alloc((size_t)B_ * H1_ * 2)};
  bf16_t* h2b[2] = {(bf16_t*)alloc((size_t)B_ * H2_ * 2),
                    (bf16_t*)alloc((size_t)B_ * H2_ * 2)};
  float* b1f = (float*)alloc(H1_ * 4);
  float* b2f = (float*)alloc(H2_ * 4);
  (void)ws_size; (void)in_sizes; (void)n_in; (void)out_size;

  auto cvt = [&](const float* src, long lds, bf16_t* dst, long ldd, int nrows,
                 int ncols) {
    dim3 g((ncols + 255) / 256, nrows);
    hipLaunchKernelGGL(cvt_rows_kernel, g, dim3(256), 0, stream, src, lds, dst,
                       ldd, ncols);
  };
  auto gemm = [&](const bf16_t* A0, long lda0, int K0, const bf16_t* A1,
                  long lda1, int K1, const bf16_t* W, long ldw,
                  const float* bias, bf16_t* Dbf, long lddb, float* Df,
                  long lddf, int N, int do_tanh) {
    dim3 g(N / 128, B_ / 64);
    hipLaunchKernelGGL(gemm_bf16_kernel, g, dim3(256), 0, stream, A0, lda0, K0,
                       A1, lda1, K1, W, ldw, bias, Dbf, lddb, Df, lddf,
                       do_tanh);
  };

  // ---- setup: weight concat/convert, bias sums, initial hidden states ----
  cvt(Wih1, I_,  W1cat,       K1CAT, H1_, I_);   // [Wih1 | Whh1] -> [H1,3072]
  cvt(Whh1, H1_, W1cat + I_,  K1CAT, H1_, H1_);
  cvt(Wih2, H1_, W2cat,       K2CAT, H2_, H1_);  // [Wih2 | Whh2] -> [H2,4096]
  cvt(Whh2, H2_, W2cat + H1_, K2CAT, H2_, H2_);
  cvt(Wo,   H2_, Wocat,       H2_,   I_,  H2_);
  cvt(h1_0, H1_, h1b[0],      H1_,   B_,  H1_);
  cvt(h2_0, H2_, h2b[0],      H2_,   B_,  H2_);
  hipLaunchKernelGGL(add_bias_kernel, dim3((H1_ + 255) / 256), dim3(256), 0,
                     stream, bih1, bhh1, b1f, H1_);
  hipLaunchKernelGGL(add_bias_kernel, dim3((H2_ + 255) / 256), dim3(256), 0,
                     stream, bih2, bhh2, b2f, H2_);

  int c1 = 0, c2 = 0;

  // ---- phase 1: 90 teacher-forced steps ----
  for (int t = 0; t < T_; ++t) {
    cvt(x + (size_t)t * I_, (long)T_ * I_, xtb, I_, B_, I_);
    gemm(xtb, I_, I_, h1b[c1], H1_, H1_, W1cat, K1CAT, b1f, h1b[1 - c1], H1_,
         nullptr, 0, H1_, 1);
    gemm(h1b[1 - c1], H1_, H1_, h2b[c2], H2_, H2_, W2cat, K2CAT, b2f,
         h2b[1 - c2], H2_, nullptr, 0, H2_, 1);
    c1 ^= 1;
    c2 ^= 1;
  }
  // out = h2 @ Wo^T + bo  (feeds phase 2; not recorded)
  gemm(h2b[c2], H2_, H2_, h2b[c2], H2_, 0, Wocat, H2_, bo, outbf, I_, nullptr,
       0, I_, 0);

  // ---- phase 2: 10 autoregressive steps ----
  for (int s = 0; s < 10; ++s) {
    float* df1 = (s == 9) ? out + h1_off : nullptr;
    float* df2 = (s == 9) ? out + h2_off : nullptr;
    gemm(outbf, I_, I_, h1b[c1], H1_, H1_, W1cat, K1CAT, b1f, h1b[1 - c1], H1_,
         df1, H1_, H1_, 1);
    gemm(h1b[1 - c1], H1_, H1_, h2b[c2], H2_, H2_, W2cat, K2CAT, b2f,
         h2b[1 - c2], H2_, df2, H2_, H2_, 1);
    c1 ^= 1;
    c2 ^= 1;
    // outs[s] = h2 @ Wo^T + bo  (recorded into d_out with row stride 10*I)
    gemm(h2b[c2], H2_, H2_, h2b[c2], H2_, 0, Wocat, H2_, bo, outbf, I_,
         out + (size_t)s * I_, 10L * I_, I_, 0);
  }
}
